// DeepGO2Model_49443663512146
// MI455X (gfx1250) — compile-verified, matrix operand order checked by reading, not run on hardware
//
#include <hip/hip_runtime.h>

// ---------------------------------------------------------------------------
// Problem constants (match reference)
// ---------------------------------------------------------------------------
#define NN    10000      // nodes
#define EE    320000     // edges
#define INF_  2560       // input feature dim
#define HH    1024       // hidden dim
#define GG    10000      // GO terms
#define RR    10         // rel_embed[R] = hasFunc
#define GPAD  10112      // 79*128, padded N for final GEMM

typedef __attribute__((ext_vector_type(16))) __bf16 v16bf;
typedef __attribute__((ext_vector_type(8)))  float  v8f;
typedef __attribute__((ext_vector_type(4)))  int    i32x4;

typedef __attribute__((address_space(1))) i32x4 as1_i32x4;
typedef __attribute__((address_space(3))) i32x4 as3_i32x4;

// ---------------------------------------------------------------------------
// Helpers
// ---------------------------------------------------------------------------
__device__ __forceinline__ unsigned cvt_bf16_rne(float f) {
    unsigned u = __float_as_uint(f);
    unsigned r = u + 0x7fffu + ((u >> 16) & 1u);
    return r >> 16;
}
// monotone float<->int map for atomicMax on floats
__device__ __forceinline__ int   enc_f(float f) { int s = __float_as_int(f); return s >= 0 ? s : (s ^ 0x7fffffff); }
__device__ __forceinline__ float dec_f(int s)   { return __int_as_float(s >= 0 ? s : (s ^ 0x7fffffff)); }

// 16B global -> LDS async copy (ASYNCcnt-tracked, gfx1250)
__device__ __forceinline__ void async_copy_b128(void* lds, const void* gsrc) {
#if __has_builtin(__builtin_amdgcn_global_load_async_to_lds_b128)
    __builtin_amdgcn_global_load_async_to_lds_b128(
        (as1_i32x4*)(unsigned long long)(size_t)gsrc,
        (as3_i32x4*)(unsigned)(size_t)lds, 0, 0);
#else
    unsigned lofs = (unsigned)(size_t)lds;           // low 32 bits = LDS offset
    asm volatile("global_load_async_to_lds_b128 %0, %1, off"
                 :: "v"(lofs), "v"(gsrc) : "memory");
#endif
}
#if __has_builtin(__builtin_amdgcn_s_wait_asynccnt)
#define WAIT_ASYNC(n) __builtin_amdgcn_s_wait_asynccnt(n)
#else
#define WAIT_ASYNC(n) asm volatile("s_wait_asynccnt %0" :: "i"(n) : "memory")
#endif

// ---------------------------------------------------------------------------
// WMMA GEMM: C[M,N] = epi(A @ B^T_layout + bias)
//   A: bf16, m-major, row stride lda (elements), K contiguous
//   B: bf16, n-major, row stride ldb (elements), K contiguous (pre-transposed)
// Block: 256 threads = 8 waves, tile 128x128, K-step 32, double-buffered LDS
// staged entirely with global_load_async_to_lds_b128.
// Wave (wm,wn) computes 64x32 = 4x2 WMMA tiles of 16x16.
// EPI: 0 = relu(v+bias[n]) -> bf16 Cb ; 1 = v -> f32 Cf ;
//      2 = sigmoid(v+|bias[n]|) -> f32 Cf (nontemporal)
// ---------------------------------------------------------------------------
template<int EPI>
__global__ __launch_bounds__(256)
void gemm_bf16_wmma(const unsigned short* __restrict__ A, int lda,
                    const unsigned short* __restrict__ B, int ldb,
                    float* __restrict__ Cf, unsigned short* __restrict__ Cb,
                    int ldc, int M, int Nn, int K,
                    const float* __restrict__ bias)
{
    // [buf][row][k] : row stride 40 bf16 (80B) keeps 16B chunks aligned
    __shared__ unsigned short As[2 * 128 * 40];
    __shared__ unsigned short Bs[2 * 128 * 40];

    const int tid   = threadIdx.x;
    const int lane  = tid & 31;
    const int wave  = tid >> 5;
    const int mWave = (wave & 1) * 64;
    const int nWave = (wave >> 1) * 32;
    const int lrow  = lane & 15;
    const int khalf = (lane < 16) ? 0 : 8;

    const int mBlk = blockIdx.y * 128;
    const int nBlk = blockIdx.x * 128;

    // this thread's two 16B chunks per tile: chunk = tid + s*256
    // row = chunk>>2 (128 rows), c = chunk&3 (4 x 16B per 32-k row)
    const int ktiles = K >> 5;

    auto issue_stage = [&](int buf, int kt) {
        #pragma unroll
        for (int s = 0; s < 2; ++s) {
            int chunk = tid + s * 256;
            int row   = chunk >> 2;
            int c8    = (chunk & 3) * 8;                 // k offset in elements
            int ga    = mBlk + row; if (ga >= M) ga = M - 1;     // clamp (rows unused)
            async_copy_b128(&As[buf * 5120 + row * 40 + c8],
                            &A[(size_t)ga * lda + kt + c8]);
            int gb    = nBlk + row;                      // B rows are padded/exact
            async_copy_b128(&Bs[buf * 5120 + row * 40 + c8],
                            &B[(size_t)gb * ldb + kt + c8]);
        }
    };

    v8f acc[4][2] = {};

    issue_stage(0, 0);
    for (int i = 0; i < ktiles; ++i) {
        const int buf = i & 1;
        if (i + 1 < ktiles) {
            issue_stage(buf ^ 1, (i + 1) << 5);
            WAIT_ASYNC(4);               // stage i done (per-wave in-order), i+1 in flight
        } else {
            WAIT_ASYNC(0);
        }
        __syncthreads();                 // stage i visible to all waves

        union Frag { v16bf v; uint4 q[2]; };
        Frag af[4], bf[2];
        const unsigned short* as = &As[buf * 5120];
        const unsigned short* bs = &Bs[buf * 5120];
        #pragma unroll
        for (int x = 0; x < 4; ++x) {
            int row = mWave + x * 16 + lrow;
            af[x].q[0] = *reinterpret_cast<const uint4*>(&as[row * 40 + khalf]);
            af[x].q[1] = *reinterpret_cast<const uint4*>(&as[row * 40 + 16 + khalf]);
        }
        #pragma unroll
        for (int y = 0; y < 2; ++y) {
            int col = nWave + y * 16 + lrow;
            bf[y].q[0] = *reinterpret_cast<const uint4*>(&bs[col * 40 + khalf]);
            bf[y].q[1] = *reinterpret_cast<const uint4*>(&bs[col * 40 + 16 + khalf]);
        }
        #pragma unroll
        for (int x = 0; x < 4; ++x)
            #pragma unroll
            for (int y = 0; y < 2; ++y)
                acc[x][y] = __builtin_amdgcn_wmma_f32_16x16x32_bf16(
                    false, af[x].v, false, bf[y].v, (short)0, acc[x][y], false, false);

        __syncthreads();                 // all reads done before buffer reuse
    }

    // epilogue: lanes 0-15 -> M=r, N=lane ; lanes 16-31 -> M=r+8, N=lane-16
    const int mOfs = (lane < 16) ? 0 : 8;
    #pragma unroll
    for (int x = 0; x < 4; ++x) {
        #pragma unroll
        for (int y = 0; y < 2; ++y) {
            int nn = nBlk + nWave + y * 16 + lrow;
            #pragma unroll
            for (int r = 0; r < 8; ++r) {
                int mm = mBlk + mWave + x * 16 + r + mOfs;
                if (mm < M && nn < Nn) {
                    float v = acc[x][y][r];
                    if (EPI == 0) {
                        v += bias[nn]; v = v > 0.f ? v : 0.f;
                        Cb[(size_t)mm * ldc + nn] = (unsigned short)cvt_bf16_rne(v);
                    } else if (EPI == 1) {
                        Cf[(size_t)mm * ldc + nn] = v;
                    } else {
                        v += fabsf(bias[nn]);
                        v = 1.f / (1.f + __expf(-v));
                        __builtin_nontemporal_store(v, &Cf[(size_t)mm * ldc + nn]);
                    }
                }
            }
        }
    }
}

// ---------------------------------------------------------------------------
// prepass: f32 -> bf16 elementwise
// ---------------------------------------------------------------------------
__global__ void k_cvt_bf16(const float* __restrict__ in,
                           unsigned short* __restrict__ out, long n)
{
    long i = (long)blockIdx.x * blockDim.x + threadIdx.x;
    if (i < n) out[i] = (unsigned short)cvt_bf16_rne(in[i]);
}

// prepass: transpose K x N (row-major) -> N x K bf16
__global__ void k_transpose_bf16(const float* __restrict__ in,
                                 unsigned short* __restrict__ out, int K, int Nn)
{
    long i = (long)blockIdx.x * blockDim.x + threadIdx.x;
    if (i >= (long)K * Nn) return;
    int n = (int)(i / K), k = (int)(i % K);
    out[(size_t)n * K + k] = (unsigned short)cvt_bf16_rne(in[(size_t)k * Nn + n]);
}

// prepass: hasFuncGO (n-major, k-contiguous, padded):
//   T[n*HH+k] = bf16(go_embed[n,k] + rel_embed[R,k]) ; zeros for n >= G
__global__ void k_hfgo(const float* __restrict__ go_embed,
                       const float* __restrict__ rel_embed,
                       unsigned short* __restrict__ T)
{
    long i = (long)blockIdx.x * blockDim.x + threadIdx.x;
    if (i >= (long)GPAD * HH) return;
    int n = (int)(i / HH), k = (int)(i % HH);
    float v = 0.f;
    if (n < GG) v = go_embed[(size_t)n * HH + k] + rel_embed[(size_t)RR * HH + k];
    T[i] = (unsigned short)cvt_bf16_rne(v);
}

// ---------------------------------------------------------------------------
// zero agg / z, init segment-max encodings
// ---------------------------------------------------------------------------
__global__ void init_buffers(float* __restrict__ agg, float* __restrict__ z,
                             int* __restrict__ menc)
{
    long i = (long)blockIdx.x * blockDim.x + threadIdx.x;
    if (i < (long)NN * HH) agg[i] = 0.f;
    if (i < NN) { z[i] = 0.f; menc[i] = 0x80000000; }
}

// ---------------------------------------------------------------------------
// el = h @ attn_l, er = h @ attn_r  (one block per row, fused)
// ---------------------------------------------------------------------------
__global__ __launch_bounds__(256)
void attn_dots(const float* __restrict__ h, const float* __restrict__ al,
               const float* __restrict__ ar, float* __restrict__ el,
               float* __restrict__ er)
{
    __shared__ float sl[256], sr[256];
    int row = blockIdx.x, t = threadIdx.x;
    const float* hr = &h[(size_t)row * HH];
    float pl = 0.f, pr = 0.f;
    for (int c = t; c < HH; c += 256) {
        float v = hr[c];
        pl += v * al[c];
        pr += v * ar[c];
    }
    sl[t] = pl; sr[t] = pr;
    __syncthreads();
    for (int s = 128; s > 0; s >>= 1) {
        if (t < s) { sl[t] += sl[t + s]; sr[t] += sr[t + s]; }
        __syncthreads();
    }
    if (t == 0) { el[row] = sl[0]; er[row] = sr[0]; }
}

// ---------------------------------------------------------------------------
// edge pass 1: e = leaky_relu(el[src]+er[dst]); segment max via atomicMax
// ---------------------------------------------------------------------------
__global__ void edge_logits(const int* __restrict__ src, const int* __restrict__ dst,
                            const float* __restrict__ el, const float* __restrict__ er,
                            float* __restrict__ e, int* __restrict__ menc)
{
    int i = blockIdx.x * blockDim.x + threadIdx.x;
    if (i >= EE) return;
    float v = el[src[i]] + er[dst[i]];
    v = v > 0.f ? v : 0.2f * v;
    e[i] = v;
    atomicMax(&menc[dst[i]], enc_f(v));
}

// ---------------------------------------------------------------------------
// edge pass 2: ee = exp(e - m[dst]); z[dst] += ee
// ---------------------------------------------------------------------------
__global__ void edge_exp(const int* __restrict__ dst, const float* __restrict__ e,
                         const int* __restrict__ menc, float* __restrict__ ee,
                         float* __restrict__ z)
{
    int i = blockIdx.x * blockDim.x + threadIdx.x;
    if (i >= EE) return;
    float v = __expf(e[i] - dec_f(menc[dst[i]]));
    ee[i] = v;
    atomicAdd(&z[dst[i]], v);
}

// ---------------------------------------------------------------------------
// aggregation: agg[dst] += (ee/z[dst]) * h[src]   (one block per edge)
// ---------------------------------------------------------------------------
__global__ __launch_bounds__(256)
void edge_agg(const int* __restrict__ src, const int* __restrict__ dst,
              const float* __restrict__ ee, const float* __restrict__ z,
              const float* __restrict__ h, float* __restrict__ agg)
{
    int eidx = blockIdx.x;
    int s = src[eidx], d = dst[eidx];
    float coef = ee[eidx] / z[d];
    const float* hs = &h[(size_t)s * HH];
    float* ad = &agg[(size_t)d * HH];
    for (int c = threadIdx.x; c < HH; c += 256)
        atomicAdd(&ad[c], coef * hs[c]);
}

// ---------------------------------------------------------------------------
// xg = bf16(agg + gat_bias)
// ---------------------------------------------------------------------------
__global__ void add_bias(const float* __restrict__ agg, const float* __restrict__ gb,
                         unsigned short* __restrict__ xg)
{
    long i = (long)blockIdx.x * blockDim.x + threadIdx.x;
    if (i >= (long)NN * HH) return;
    xg[i] = (unsigned short)cvt_bf16_rne(agg[i] + gb[i % HH]);
}

// ---------------------------------------------------------------------------
// launch
// ---------------------------------------------------------------------------
extern "C" void kernel_launch(void* const* d_in, const int* in_sizes, int n_in,
                              void* d_out, int out_size, void* d_ws, size_t ws_size,
                              hipStream_t stream)
{
    const float* features = (const float*)d_in[0];
    const int*   src      = (const int*)  d_in[1];
    const int*   dst      = (const int*)  d_in[2];
    const float* W1       = (const float*)d_in[3];
    const float* b1       = (const float*)d_in[4];
    const float* fc_w     = (const float*)d_in[5];
    const float* attn_l   = (const float*)d_in[6];
    const float* attn_r   = (const float*)d_in[7];
    const float* gat_bias = (const float*)d_in[8];
    const float* go_embed = (const float*)d_in[9];
    const float* go_rad   = (const float*)d_in[10];
    const float* rel_embed= (const float*)d_in[11];
    float* out = (float*)d_out;

    // workspace carve-up (256B aligned)
    char* ws = (char*)d_ws;
    size_t off = 0;
    auto alloc = [&](size_t bytes) -> char* {
        char* p = ws + off;
        off += (bytes + 255) & ~(size_t)255;
        return p;
    };
    unsigned short* feat_bf = (unsigned short*)alloc((size_t)NN * INF_ * 2);
    unsigned short* W1t     = (unsigned short*)alloc((size_t)INF_ * HH * 2);
    unsigned short* fct     = (unsigned short*)alloc((size_t)HH * HH * 2);
    unsigned short* hfgo    = (unsigned short*)alloc((size_t)GPAD * HH * 2);
    unsigned short* x_bf    = (unsigned short*)alloc((size_t)NN * HH * 2);
    unsigned short* xg_bf   = (unsigned short*)alloc((size_t)NN * HH * 2);
    float* h     = (float*)alloc((size_t)NN * HH * 4);
    float* agg   = (float*)alloc((size_t)NN * HH * 4);
    float* el    = (float*)alloc((size_t)NN * 4);
    float* er    = (float*)alloc((size_t)NN * 4);
    float* e     = (float*)alloc((size_t)EE * 4);
    float* ee    = (float*)alloc((size_t)EE * 4);
    float* z     = (float*)alloc((size_t)NN * 4);
    int*   menc  = (int*)  alloc((size_t)NN * 4);

    const int T = 256;
    // prepasses (independent)
    {
        long nf = (long)NN * INF_;
        k_cvt_bf16<<<dim3((nf + T - 1) / T), dim3(T), 0, stream>>>(features, feat_bf, nf);
        long nw = (long)INF_ * HH;
        k_transpose_bf16<<<dim3((nw + T - 1) / T), dim3(T), 0, stream>>>(W1, W1t, INF_, HH);
        long nc = (long)HH * HH;
        k_transpose_bf16<<<dim3((nc + T - 1) / T), dim3(T), 0, stream>>>(fc_w, fct, HH, HH);
        long ng = (long)GPAD * HH;
        k_hfgo<<<dim3((ng + T - 1) / T), dim3(T), 0, stream>>>(go_embed, rel_embed, hfgo);
        long tot = (long)NN * HH;
        init_buffers<<<dim3((tot + T - 1) / T), dim3(T), 0, stream>>>(agg, z, menc);
    }

    const int MT = (NN + 127) / 128;   // 79
    // GEMM1: x = relu(features @ W1 + b1)  -> bf16
    gemm_bf16_wmma<0><<<dim3(HH / 128, MT), dim3(T), 0, stream>>>(
        feat_bf, INF_, W1t, INF_, nullptr, x_bf, HH, NN, HH, INF_, b1);
    // GEMM2: h = x @ fc_w  -> f32
    gemm_bf16_wmma<1><<<dim3(HH / 128, MT), dim3(T), 0, stream>>>(
        x_bf, HH, fct, HH, h, nullptr, HH, NN, HH, HH, nullptr);
    // el/er
    attn_dots<<<dim3(NN), dim3(T), 0, stream>>>(h, attn_l, attn_r, el, er);
    // edge softmax + aggregation
    edge_logits<<<dim3((EE + T - 1) / T), dim3(T), 0, stream>>>(src, dst, el, er, e, menc);
    edge_exp<<<dim3((EE + T - 1) / T), dim3(T), 0, stream>>>(dst, e, menc, ee, z);
    edge_agg<<<dim3(EE), dim3(T), 0, stream>>>(src, dst, ee, z, h, agg);
    // xg = agg + gat_bias -> bf16
    {
        long tot = (long)NN * HH;
        add_bias<<<dim3((tot + T - 1) / T), dim3(T), 0, stream>>>(agg, gat_bias, xg_bf);
    }
    // GEMM3: out = sigmoid(xg @ hasFuncGO^T + |rad|)
    gemm_bf16_wmma<2><<<dim3(GPAD / 128, MT), dim3(T), 0, stream>>>(
        xg_bf, HH, hfgo, HH, out, nullptr, GG, NN, GG, HH, go_rad);
}
